// HDRNet_8375186227447
// MI455X (gfx1250) — matrix operand order repaired and checked by reading, not compiled, not run on hardware
//
#include <hip/hip_runtime.h>
#include <hip/hip_bf16.h>

typedef __attribute__((ext_vector_type(16))) _Float16 v16h;
typedef __attribute__((ext_vector_type(8)))  float    v8f;

// ---------------------------------------------------------------------------
// Implicit-GEMM conv (and FC via KHW=1) using v_wmma_f32_16x16x32_f16.
// One wave (32 threads) computes a 16-pixel x 16-outchannel tile.
//   A[m][k] : m = output pixel (tile row), k = cin*KHW*KHW + ky*KHW + kx
//   B[k][n] : n = output channel
// Fragment layouts follow CDNA5 ISA 7.12.2 (16-bit A 16x32, B 32x16, f32 C/D).
// ---------------------------------------------------------------------------
template<int KHW>
__global__ void __launch_bounds__(32)
conv_wmma_kernel(const float* __restrict__ in, const float* __restrict__ w,
                 const float* __restrict__ bias, const float* __restrict__ fuse,
                 float* __restrict__ out,
                 int Nimg, int Cin, int Cout, int Hin, int Win,
                 int Hout, int Wout, int stride, int pad, int do_relu)
{
    constexpr int KHW2 = KHW * KHW;
    const int lane = (int)(threadIdx.x & 31u);
    const int half = lane >> 4;          // 0: lanes 0-15, 1: lanes 16-31
    const int lm   = lane & 15;
    const int HW   = Hout * Wout;
    const int npix = Nimg * HW;
    const int Kin  = Cin * KHW2;

    // Pixel for this lane's A rows (A: M = lane&15 for both halves)
    const int  pa    = (int)blockIdx.x * 16 + lm;
    const bool pa_ok = pa < npix;
    const int  na    = pa_ok ? pa / HW : 0;
    const int  rema  = pa_ok ? pa - na * HW : 0;
    const int  oya   = rema / Wout;
    const int  oxa   = rema - oya * Wout;

    // Output channel for this lane's B column / D column
    const int  chb    = (int)blockIdx.y * 16 + lm;
    const bool chb_ok = chb < Cout;

    auto fetchA = [&](int k) -> float {
        if (!pa_ok || k >= Kin) return 0.0f;
        int cin = k / KHW2;
        int r   = k - cin * KHW2;
        int ky  = r / KHW;
        int kx  = r - ky * KHW;
        int iy  = oya * stride - pad + ky;
        int ix  = oxa * stride - pad + kx;
        if (iy < 0 || iy >= Hin || ix < 0 || ix >= Win) return 0.0f;
        float v = in[((na * Cin + cin) * Hin + iy) * Win + ix];
        if (fuse) v = fmaxf(v + fuse[na * Cin + cin], 0.0f); // fusion = relu(loc+glob)
        return v;
    };
    auto fetchB = [&](int k) -> float {
        if (!chb_ok || k >= Kin) return 0.0f;
        int cin = k / KHW2;
        int r   = k - cin * KHW2;
        return w[(chb * Cin + cin) * KHW2 + r];
    };

    v8f acc = {};
    for (int kk = 0; kk < Kin; kk += 32) {
        v16h a, b;
#pragma unroll
        for (int i = 0; i < 8; ++i) {
            // A 16x32 f16 layout: VGPR i holds K pair k0,k0+1 where
            // k0 = (i>=4?16:0) + half*8 + (i&3)*2
            int g  = i >> 2, j = i & 3;
            int ka = kk + g * 16 + half * 8 + j * 2;
            a[2 * i]     = (_Float16)fetchA(ka);
            a[2 * i + 1] = (_Float16)fetchA(ka + 1);
            // B 32x16 f16 layout: same K order as A,
            // VGPR i holds K pair (i>=4?16:0) + half*8 + (i&3)*2, +1
            int kb = kk + g * 16 + half * 8 + j * 2;
            b[2 * i]     = (_Float16)fetchB(kb);
            b[2 * i + 1] = (_Float16)fetchB(kb + 1);
        }
        acc = __builtin_amdgcn_wmma_f32_16x16x32_f16(
            /*neg_a=*/false, a, /*neg_b=*/false, b,
            /*c_mod=*/(short)0, acc, /*reuse_a=*/false, /*reuse_b=*/false);
    }

    const float bv = (bias != nullptr && chb_ok) ? bias[chb] : 0.0f;
#pragma unroll
    for (int r = 0; r < 8; ++r) {
        // D layout: lane l, VGPR r -> D[m = r + 8*(l>>4)][n = l&15]
        int m = r + 8 * half;
        int p = (int)blockIdx.x * 16 + m;
        if (p < npix && chb_ok) {
            int n   = p / HW;
            int rr  = p - n * HW;
            int oy  = rr / Wout;
            int ox  = rr - oy * Wout;
            float v = acc[r] + bv;
            if (do_relu) v = fmaxf(v, 0.0f);
            out[((n * Cout + chb) * Hout + oy) * Wout + ox] = v;
        }
    }
}

// ---------------------------------------------------------------------------
// Permute grid [N][96][16][16] (96 = 12 coeff * 8 depth) into
// [N][gy][gx][z][12] so a trilinear corner is 3 contiguous float4.
// ---------------------------------------------------------------------------
__global__ void grid_permute_kernel(const float* __restrict__ src,
                                    float* __restrict__ dst, int Nimg)
{
    int idx   = (int)(blockIdx.x * blockDim.x + threadIdx.x);
    int total = Nimg * 96 * 256;
    if (idx >= total) return;
    int n   = idx / (96 * 256);
    int rem = idx - n * 96 * 256;
    int cc  = rem / 256;               // 0..95 = c*8 + d
    int yx  = rem - cc * 256;
    int gy  = yx / 16, gx = yx - gy * 16;
    int c   = cc / 8,  d  = cc - c * 8;
    dst[(((n * 16 + gy) * 16 + gx) * 8 + d) * 12 + c] = src[idx];
}

// ---------------------------------------------------------------------------
// Fused full-res pass: guide curve + trilinear slice + per-pixel affine apply.
// Bandwidth-bound; pure VALU + vectorized (b128) cached grid gathers.
// ---------------------------------------------------------------------------
__global__ void __launch_bounds__(256)
fullres_kernel(const float* __restrict__ img, const float* __restrict__ gridP,
               const float* __restrict__ M, const float* __restrict__ Mb,
               const float* __restrict__ thr, const float* __restrict__ slopes,
               const float* __restrict__ gbias, float* __restrict__ out,
               int Nimg, int H, int W)
{
    int idx = (int)(blockIdx.x * blockDim.x + threadIdx.x);
    int HW  = H * W;
    if (idx >= Nimg * HW) return;
    int n   = idx / HW;
    int rem = idx - n * HW;
    int y   = rem / W;
    int x   = rem - y * W;

    const float fr = img[(n * 3 + 0) * HW + rem];
    const float fg = img[(n * 3 + 1) * HW + rem];
    const float fb = img[(n * 3 + 2) * HW + rem];

    // Guide: g = f @ M + Mb; piecewise-linear per channel; mean + bias; clip.
    float s = 0.0f;
#pragma unroll
    for (int j = 0; j < 3; ++j) {
        float g = fr * M[0 * 3 + j] + fg * M[1 * 3 + j] + fb * M[2 * 3 + j] + Mb[j];
        float sj = 0.0f;
#pragma unroll
        for (int t = 0; t < 16; ++t)
            sj += slopes[j * 16 + t] * fmaxf(g - thr[j * 16 + t], 0.0f);
        s += sj;
    }
    float guide = s * (1.0f / 3.0f) + gbias[0];
    guide = fminf(fmaxf(guide, 0.0f), 1.0f);

    // unnorm(c,size) = clip(((c+1)*size-1)*0.5, 0, size-1) with c=linspace(-1,1)
    float ix = fminf(fmaxf(16.0f * (float)x / (float)(W - 1) - 0.5f, 0.0f), 15.0f);
    float iy = fminf(fmaxf(16.0f * (float)y / (float)(H - 1) - 0.5f, 0.0f), 15.0f);
    float iz = fminf(fmaxf(8.0f * guide - 0.5f, 0.0f), 7.0f);
    int x0 = (int)floorf(ix); float wx = ix - (float)x0; int x1 = (x0 + 1 < 15) ? x0 + 1 : 15;
    int y0 = (int)floorf(iy); float wy = iy - (float)y0; int y1 = (y0 + 1 < 15) ? y0 + 1 : 15;
    int z0 = (int)floorf(iz); float wz = iz - (float)z0; int z1 = (z0 + 1 < 7)  ? z0 + 1 : 7;

    float acc[12];
#pragma unroll
    for (int c = 0; c < 12; ++c) acc[c] = 0.0f;

    const float wcy[2] = {1.0f - wy, wy};
    const float wcx[2] = {1.0f - wx, wx};
    const int   ys[2]  = {y0, y1};
    const int   xs[2]  = {x0, x1};

#pragma unroll
    for (int cy = 0; cy < 2; ++cy) {
#pragma unroll
        for (int cx = 0; cx < 2; ++cx) {
            float wyx = wcy[cy] * wcx[cx];
            const float* base =
                gridP + (size_t)(((n * 16 + ys[cy]) * 16 + xs[cx]) * 8) * 12;
            const float4* p0 = reinterpret_cast<const float4*>(base + (size_t)z0 * 12);
            const float4* p1 = reinterpret_cast<const float4*>(base + (size_t)z1 * 12);
            float4 A0 = p0[0], A1 = p0[1], A2 = p0[2];
            float4 B0 = p1[0], B1 = p1[1], B2 = p1[2];
            float ga[12] = {A0.x, A0.y, A0.z, A0.w, A1.x, A1.y, A1.z, A1.w,
                            A2.x, A2.y, A2.z, A2.w};
            float gb[12] = {B0.x, B0.y, B0.z, B0.w, B1.x, B1.y, B1.z, B1.w,
                            B2.x, B2.y, B2.z, B2.w};
#pragma unroll
            for (int c = 0; c < 12; ++c)
                acc[c] += wyx * (ga[c] + wz * (gb[c] - ga[c]));
        }
    }

    float rr = fr * acc[0] + fg * acc[1] + fb * acc[2]  + acc[3];
    float gg = fr * acc[4] + fg * acc[5] + fb * acc[6]  + acc[7];
    float bb = fr * acc[8] + fg * acc[9] + fb * acc[10] + acc[11];
    out[(n * 3 + 0) * HW + rem] = rr;
    out[(n * 3 + 1) * HW + rem] = gg;
    out[(n * 3 + 2) * HW + rem] = bb;
}

// ---------------------------------------------------------------------------

extern "C" void kernel_launch(void* const* d_in, const int* in_sizes, int n_in,
                              void* d_out, int out_size, void* d_ws, size_t ws_size,
                              hipStream_t stream)
{
    (void)in_sizes; (void)n_in; (void)out_size; (void)ws_size;

    const float* lowres  = (const float*)d_in[0];
    const float* fullres = (const float*)d_in[1];
    const float* sw1 = (const float*)d_in[2];  const float* sb1 = (const float*)d_in[3];
    const float* sw2 = (const float*)d_in[4];  const float* sb2 = (const float*)d_in[5];
    const float* sw3 = (const float*)d_in[6];  const float* sb3 = (const float*)d_in[7];
    const float* sw4 = (const float*)d_in[8];  const float* sb4 = (const float*)d_in[9];
    const float* gw1 = (const float*)d_in[10]; const float* gb1 = (const float*)d_in[11];
    const float* gw2 = (const float*)d_in[12]; const float* gb2 = (const float*)d_in[13];
    const float* fw3 = (const float*)d_in[14]; const float* fb3 = (const float*)d_in[15];
    const float* fw4 = (const float*)d_in[16]; const float* fb4 = (const float*)d_in[17];
    const float* fw5 = (const float*)d_in[18]; const float* fb5 = (const float*)d_in[19];
    const float* lw1 = (const float*)d_in[20]; const float* lb1 = (const float*)d_in[21];
    const float* lw2 = (const float*)d_in[22];
    const float* pw  = (const float*)d_in[23]; const float* pb  = (const float*)d_in[24];
    const float* Mm  = (const float*)d_in[25]; const float* Mb  = (const float*)d_in[26];
    const float* thr = (const float*)d_in[27]; const float* slo = (const float*)d_in[28];
    const float* gbias = (const float*)d_in[29];
    float* out = (float*)d_out;

    // Workspace layout (f32 elements)
    float* ws    = (float*)d_ws;
    float* act1  = ws;                  // [2,8,128,128]  262144
    float* act2  = act1  + 262144;      // [2,16,64,64]   131072
    float* act3  = act2  + 131072;      // [2,32,32,32]    65536
    float* splat = act3  + 65536;       // [2,64,16,16]    32768
    float* g1b   = splat + 32768;       // [2,64,8,8]       8192
    float* g2b   = g1b   + 8192;        // [2,64,4,4]       2048
    float* fc3b  = g2b   + 2048;        // [2,256]           512
    float* fc4b  = fc3b  + 512;         // [2,128]           256
    float* glob  = fc4b  + 256;         // [2,64]            128
    float* loc1  = glob  + 128;         // [2,64,16,16]    32768
    float* loc2  = loc1  + 32768;       // [2,64,16,16]    32768
    float* gridb = loc2  + 32768;       // [2,96,16,16]    49152
    float* gridP = gridb + 49152;       // [2,16,16,8,12]  49152

    const dim3 wv(32);
    // --- low-res coefficient CNN (WMMA implicit GEMM) ---
    conv_wmma_kernel<3><<<dim3(2048, 1), wv, 0, stream>>>(lowres, sw1, sb1, nullptr, act1,  2,   3,   8, 256, 256, 128, 128, 2, 1, 1);
    conv_wmma_kernel<3><<<dim3(512,  1), wv, 0, stream>>>(act1,   sw2, sb2, nullptr, act2,  2,   8,  16, 128, 128,  64,  64, 2, 1, 1);
    conv_wmma_kernel<3><<<dim3(128,  2), wv, 0, stream>>>(act2,   sw3, sb3, nullptr, act3,  2,  16,  32,  64,  64,  32,  32, 2, 1, 1);
    conv_wmma_kernel<3><<<dim3(32,   4), wv, 0, stream>>>(act3,   sw4, sb4, nullptr, splat, 2,  32,  64,  32,  32,  16,  16, 2, 1, 1);
    conv_wmma_kernel<3><<<dim3(8,    4), wv, 0, stream>>>(splat,  gw1, gb1, nullptr, g1b,   2,  64,  64,  16,  16,   8,   8, 2, 1, 1);
    conv_wmma_kernel<3><<<dim3(2,    4), wv, 0, stream>>>(g1b,    gw2, gb2, nullptr, g2b,   2,  64,  64,   8,   8,   4,   4, 2, 1, 1);
    // --- FC layers as 1x1 "convs" with H=W=1 ---
    conv_wmma_kernel<1><<<dim3(1,   16), wv, 0, stream>>>(g2b,    fw3, fb3, nullptr, fc3b,  2, 1024, 256,   1,   1,   1,   1, 1, 0, 1);
    conv_wmma_kernel<1><<<dim3(1,    8), wv, 0, stream>>>(fc3b,   fw4, fb4, nullptr, fc4b,  2,  256, 128,   1,   1,   1,   1, 1, 0, 1);
    conv_wmma_kernel<1><<<dim3(1,    4), wv, 0, stream>>>(fc4b,   fw5, fb5, nullptr, glob,  2,  128,  64,   1,   1,   1,   1, 1, 0, 0);
    // --- local path ---
    conv_wmma_kernel<3><<<dim3(32,   4), wv, 0, stream>>>(splat,  lw1, lb1, nullptr, loc1,  2,  64,  64,  16,  16,  16,  16, 1, 1, 1);
    conv_wmma_kernel<3><<<dim3(32,   4), wv, 0, stream>>>(loc1,   lw2, nullptr, nullptr, loc2, 2, 64, 64,  16,  16,  16,  16, 1, 1, 0);
    // --- fusion (relu(loc+glob) fused into A-fetch) + 1x1 projection -> grid ---
    conv_wmma_kernel<1><<<dim3(32,   6), wv, 0, stream>>>(loc2,   pw,  pb,  glob,    gridb, 2,  64,  96,  16,  16,  16,  16, 1, 0, 0);

    // --- permute grid for vectorized slicing ---
    grid_permute_kernel<<<dim3(192), dim3(256), 0, stream>>>(gridb, gridP, 2);

    // --- fused guide + trilinear slice + apply over full-res image ---
    fullres_kernel<<<dim3(8192), dim3(256), 0, stream>>>(
        fullres, gridP, Mm, Mb, thr, slo, gbias, out, 2, 1024, 1024);
}
